// Cfconv_53858889891846
// MI455X (gfx1250) — matrix operand order repaired and checked by reading, not compiled, not run on hardware
//
#include <hip/hip_runtime.h>
#include <hip/hip_bf16.h>

typedef _Float16 v16h __attribute__((ext_vector_type(16)));
typedef _Float16 v8h  __attribute__((ext_vector_type(8)));
typedef _Float16 v2h  __attribute__((ext_vector_type(2)));
typedef float    v8f  __attribute__((ext_vector_type(8)));

#define N_RBF   300
#define KPAD    320          // K padded to multiple of 32 for 16x16x32 WMMA
#define GAMMA   10.0f
#define LOG2E   1.4426950408889634f
#define LN2     0.69314718055994530942f
#define BDIM    16
#define NDIM    128
#define ITERS   8            // (b,i) rows handled per workgroup

struct H16 { v2h p[8]; };    // 8 packed f16 pairs == one v16h WMMA operand

// pack two f32 to one f16 pair VGPR (v_cvt_pk_f16_f32, round-toward-zero)
__device__ __forceinline__ v2h pk2(float x0, float x1) {
    return __builtin_bit_cast(v2h, __builtin_amdgcn_cvt_pkrtz(x0, x1));
}

// shifted softplus: max(v,0) + ln2*log2((1 + exp(-|v|)) / 2)
// fabs/neg are free source modifiers; exp2/log2 are single TRANS ops.
__device__ __forceinline__ float ssp_f(float v) {
    const float t = __builtin_amdgcn_exp2f(-fabsf(v) * LOG2E);        // exp(-|v|)
    return fmaf(LN2, __builtin_amdgcn_logf(fmaf(t, 0.5f, 0.5f)), fmaxf(v, 0.0f));
}

// Generate 8 RBF features exp2(A*k^2 + B*k + C) for k = kb..kb+7 into 4 packed pairs.
// Per element: 1 v_add-folded fma + 1 fma + v_exp_f32; pairs packed with v_cvt_pk_f16_f32.
__device__ __forceinline__ void gen8(float kb, float Bf, float Cf, v2h* dst) {
    const float Af = -0.01f * GAMMA * LOG2E;                 // literal
    const float P  = fmaf(kb, fmaf(kb, Af, Bf), Cf);         // arg at e=0
    const float Q  = fmaf(2.0f * Af, kb, Bf);                // d(arg)/de at e=0
    #pragma unroll
    for (int ep = 0; ep < 4; ++ep) {
        const float e0 = (float)(2 * ep), e1 = (float)(2 * ep + 1);
        const float x0 = __builtin_amdgcn_exp2f(fmaf(e0, Q, P + Af * e0 * e0));
        const float x1 = __builtin_amdgcn_exp2f(fmaf(e1, Q, P + Af * e1 * e1));
        dst[ep] = pk2(x0, x1);
    }
}

// Same, but zeroes elements with k >= N_RBF (only used in the peeled last K-step).
__device__ __forceinline__ void gen8_tail(float kb, int kbi, float Bf, float Cf, v2h* dst) {
    const float Af = -0.01f * GAMMA * LOG2E;
    const float P  = fmaf(kb, fmaf(kb, Af, Bf), Cf);
    const float Q  = fmaf(2.0f * Af, kb, Bf);
    #pragma unroll
    for (int ep = 0; ep < 4; ++ep) {
        const float e0 = (float)(2 * ep), e1 = (float)(2 * ep + 1);
        float x0 = __builtin_amdgcn_exp2f(fmaf(e0, Q, P + Af * e0 * e0));
        float x1 = __builtin_amdgcn_exp2f(fmaf(e1, Q, P + Af * e1 * e1));
        x0 = (kbi + 2 * ep     < N_RBF) ? x0 : 0.0f;
        x1 = (kbi + 2 * ep + 1 < N_RBF) ? x1 : 0.0f;
        dst[ep] = pk2(x0, x1);
    }
}

// A-matrix 16x32 f16 layout: element e of v16h <-> K = (e<8?0:16) + half*8 + (e&7)
__device__ __forceinline__ v16h load_a_frag(const _Float16* rowp, int kbase, int half) {
    const v8h lo = *(const v8h*)(rowp + kbase + half * 8);
    const v8h hi = *(const v8h*)(rowp + kbase + 16 + half * 8);
    return __builtin_shufflevector(lo, hi, 0,1,2,3,4,5,6,7,8,9,10,11,12,13,14,15);
}

// B-matrix 32x16 f16 layout: lane holds column N=lane&15,
// element e <-> K = (e<8?0:16) + half*8 + (e&7)
__device__ __forceinline__ v16h load_b_frag(const _Float16* colp, int kbase, int half) {
    const v8h lo = *(const v8h*)(colp + kbase + half * 8);
    const v8h hi = *(const v8h*)(colp + kbase + 16 + half * 8);
    return __builtin_shufflevector(lo, hi, 0,1,2,3,4,5,6,7,8,9,10,11,12,13,14,15);
}

__global__ __launch_bounds__(256)
void cfconv_wmma_kernel(const float* __restrict__ x,
                        const float* __restrict__ dist,
                        const float* __restrict__ W1,
                        const float* __restrict__ b1,
                        const float* __restrict__ W2,
                        const float* __restrict__ b2,
                        float* __restrict__ out)
{
    __shared__ __align__(16) _Float16 sW1T[64][KPAD]; // W1 transposed [f][k], 40 KB
    __shared__ __align__(16) _Float16 sW2T[64][64];   // W2 transposed [g][f],  8 KB
    __shared__ __align__(16) _Float16 sH[128][64];    // h staging [j][f],     16 KB
    __shared__ float sD[128];
    __shared__ float sS[64];

    const int tid   = threadIdx.x;
    const int wave  = tid >> 5;
    const int lane  = tid & 31;
    const int half  = lane >> 4;
    const int col   = lane & 15;
    const int jbase = wave * 16;

    // ---- Stage weights (f32 -> f16, transposed) with float4 global loads ----
    {
        const float4* W1v = (const float4*)W1;                 // [300][64] -> 4800 vec4
        for (int idx = tid; idx < (N_RBF * 64) / 4; idx += 256) {
            const float4 w = W1v[idx];
            const int k  = idx >> 4;
            const int f0 = (idx & 15) << 2;
            sW1T[f0 + 0][k] = (_Float16)w.x;
            sW1T[f0 + 1][k] = (_Float16)w.y;
            sW1T[f0 + 2][k] = (_Float16)w.z;
            sW1T[f0 + 3][k] = (_Float16)w.w;
        }
        for (int idx = tid; idx < 64 * (KPAD - N_RBF); idx += 256) {
            const int f = idx / (KPAD - N_RBF);
            const int k = N_RBF + idx % (KPAD - N_RBF);
            sW1T[f][k] = (_Float16)0.0f;
        }
        const float4* W2v = (const float4*)W2;                 // [64][64] -> 1024 vec4
        for (int idx = tid; idx < (64 * 64) / 4; idx += 256) {
            const float4 w = W2v[idx];
            const int f  = idx >> 4;
            const int g0 = (idx & 15) << 2;
            sW2T[g0 + 0][f] = (_Float16)w.x;
            sW2T[g0 + 1][f] = (_Float16)w.y;
            sW2T[g0 + 2][f] = (_Float16)w.z;
            sW2T[g0 + 3][f] = (_Float16)w.w;
        }
    }

    // Biases: per-lane constants for the whole workgroup lifetime
    float bv1[4], bv2[4];
    #pragma unroll
    for (int nt = 0; nt < 4; ++nt) {
        bv1[nt] = b1[nt * 16 + col];
        bv2[nt] = b2[nt * 16 + col];
    }

    const float h8 = (float)(half << 3);

    for (int it = 0; it < ITERS; ++it) {
        const int bi = blockIdx.x * ITERS + it;    // flattened (b*128 + i)
        __syncthreads();                           // prior iter's sS consumed; weights staged
        if (tid < 64)  sS[tid] = 0.0f;
        if (tid < 128) sD[tid] = dist[bi * NDIM + tid];
        __syncthreads();

        // ---- GEMM1: e_k[128x320] @ W1[320x64] -> h (per-wave 16-row M-tile) ----
        v8f acc[4] = {};
        const float d  = sD[jbase + col];
        // exponent(k) = Af*k^2 + Bf*k + Cf  (already scaled by log2e for v_exp_f32)
        const float Bf = (0.2f * GAMMA * LOG2E) * d;
        const float Cf = (-GAMMA * LOG2E) * d * d;

        #pragma unroll 3
        for (int ks = 0; ks < 9; ++ks) {           // k = 0..287, no masking needed
            const int kbase = ks * 32;
            H16 t;
            gen8((float)kbase + h8,          Bf, Cf, t.p);      // K group 0..7  (+half*8)
            gen8((float)(kbase + 16) + h8,   Bf, Cf, t.p + 4);  // K group 16..23(+half*8)
            const v16h a = __builtin_bit_cast(v16h, t);
            #pragma unroll
            for (int nt = 0; nt < 4; ++nt) {
                const v16h bfr = load_b_frag(&sW1T[nt * 16 + col][0], kbase, half);
                acc[nt] = __builtin_amdgcn_wmma_f32_16x16x32_f16(
                    false, a, false, bfr, (short)0, acc[nt], false, false);
            }
        }
        {   // peeled last K-step: k = 288..319, zero k >= 300
            const int kbase = 288;
            const int kb0 = kbase + (half << 3), kb1 = kbase + 16 + (half << 3);
            H16 t;
            gen8_tail((float)kb0, kb0, Bf, Cf, t.p);
            gen8_tail((float)kb1, kb1, Bf, Cf, t.p + 4);
            const v16h a = __builtin_bit_cast(v16h, t);
            #pragma unroll
            for (int nt = 0; nt < 4; ++nt) {
                const v16h bfr = load_b_frag(&sW1T[nt * 16 + col][0], kbase, half);
                acc[nt] = __builtin_amdgcn_wmma_f32_16x16x32_f16(
                    false, a, false, bfr, (short)0, acc[nt], false, false);
            }
        }

        // bias + ssp, write h tile to LDS (C layout: vgpr r, lane L -> M=r+8*(L>=16), N=L&15)
        const int mofs = half * 8;
        #pragma unroll
        for (int nt = 0; nt < 4; ++nt) {
            #pragma unroll
            for (int r = 0; r < 8; ++r) {
                sH[jbase + mofs + r][nt * 16 + col] = (_Float16)ssp_f(acc[nt][r] + bv1[nt]);
            }
        }
        // Each wave reads back only the 16 sH rows it wrote: wave-local LDS dependency,
        // drain this wave's DS stores instead of a full workgroup barrier.
        asm volatile("s_wait_dscnt 0x0" ::: "memory");

        // ---- GEMM2: h[16x64] @ W2[64x64] -> w_l tile ----
        v8f acc2[4] = {};
        #pragma unroll
        for (int ks = 0; ks < 2; ++ks) {
            const int kbase = ks * 32;
            const v16h a2 = load_a_frag(&sH[jbase + col][0], kbase, half);
            #pragma unroll
            for (int nt = 0; nt < 4; ++nt) {
                const v16h bfr = load_b_frag(&sW2T[nt * 16 + col][0], kbase, half);
                acc2[nt] = __builtin_amdgcn_wmma_f32_16x16x32_f16(
                    false, a2, false, bfr, (short)0, acc2[nt], false, false);
            }
        }

        // bias + ssp + partial column sums over this wave's 16 j rows, reduce in LDS
        #pragma unroll
        for (int nt = 0; nt < 4; ++nt) {
            float part = 0.0f;
            #pragma unroll
            for (int r = 0; r < 8; ++r) part += ssp_f(acc2[nt][r] + bv2[nt]);
            atomicAdd(&sS[nt * 16 + col], part);   // ds_add_f32
        }
        __syncthreads();

        // out[b,i,g] = x[b,i,g] * sum_j w_l[b,i,j,g]
        if (tid < 64) {
            out[bi * 64 + tid] = x[bi * 64 + tid] * sS[tid];
        }
    }
}

extern "C" void kernel_launch(void* const* d_in, const int* in_sizes, int n_in,
                              void* d_out, int out_size, void* d_ws, size_t ws_size,
                              hipStream_t stream) {
    const float* x    = (const float*)d_in[0];
    const float* dist = (const float*)d_in[1];
    const float* W1   = (const float*)d_in[2];
    const float* b1   = (const float*)d_in[3];
    const float* W2   = (const float*)d_in[4];
    const float* b2   = (const float*)d_in[5];
    float* out = (float*)d_out;

    const int n_rows = BDIM * NDIM;                  // 2048 (b,i) rows
    dim3 grid(n_rows / ITERS);                       // 256 workgroups
    cfconv_wmma_kernel<<<grid, 256, 0, stream>>>(x, dist, W1, b1, W2, b2, out);
}
